// HierarchicalMolecularGNN_70342974374330
// MI455X (gfx1250) — matrix operand order repaired and checked by reading, not compiled, not run on hardware
//
#include <hip/hip_runtime.h>
#include <cstdint>
#include <cstddef>

// ---------------- problem constants (fixed by reference) ----------------
#define NN    150000
#define EE    300000
#define ENL   (EE + NN)     // edges + self loops, loops appended after edges
#define HID   256
#define HEADS 4
#define CH    64
#define BB    4096
#define DIN   155
#define KPROJ 160           // DIN padded to multiple of 32
#define DEA   6
#define LAY   4
#define NEG_SLOPE 0.2f
#define BN_EPS    1e-5f

#define CDIV(a,b) (((a)+(b)-1)/(b))

typedef __attribute__((ext_vector_type(16))) _Float16 v16h;
typedef __attribute__((ext_vector_type(8)))  _Float16 v8h;
typedef __attribute__((ext_vector_type(8)))  float    v8f;

// ---------------- small helpers ----------------
__device__ inline unsigned f2ord(float f) {
    unsigned u = __float_as_uint(f);
    return (u & 0x80000000u) ? ~u : (u | 0x80000000u);
}
__device__ inline float ord2f(unsigned o) {
    unsigned u = (o & 0x80000000u) ? (o & 0x7fffffffu) : ~o;
    return __uint_as_float(u);
}

// ---------------- zero fill (float4 granularity) ----------------
__global__ void zero_kernel(float* __restrict__ p, int n4) {
    int i = blockIdx.x * blockDim.x + threadIdx.x;
    if (i < n4) ((float4*)p)[i] = make_float4(0.f, 0.f, 0.f, 0.f);
}

// ---------------- weight convert + transpose: W[K][256] f32 -> Wt[256][Kpad] f16 ----------------
__global__ void cvt_wt_kernel(const float* __restrict__ W, _Float16* __restrict__ Wt,
                              int K, int Kpad) {
    int idx = blockIdx.x * blockDim.x + threadIdx.x;
    if (idx >= 256 * Kpad) return;
    int n = idx / Kpad, k = idx % Kpad;
    float v = (k < K) ? W[(size_t)k * 256 + n] : 0.f;
    Wt[(size_t)n * Kpad + k] = (_Float16)v;
}

// ---------------- WMMA GEMM: C[M,256] = A[M,lda(K)] * W, f16 inputs f32 acc ----------------
// WG: 256 threads = 8 waves (2x4 wave grid), tile 64x256, K chunked by 32 through LDS.
// A is register double-buffered (HBM traffic); B staged directly (L2-resident weights).
// B fragments loaded one N-tile at a time to cap VGPR pressure (no spills).
// vec_ok: uniform flag, lda % 4 == 0 -> aligned float4 staging path.
__global__ __launch_bounds__(256)
void gemm_f16_wmma(const float* __restrict__ A, int lda, int M,
                   const _Float16* __restrict__ Wt, int K,
                   float* __restrict__ C,
                   const float* __restrict__ bias, int do_relu, int vec_ok) {
    __shared__ __align__(16) _Float16 As[64 * 40];    // [row][k] stride 40 (80B, 16B aligned frags)
    __shared__ __align__(16) _Float16 Bs[256 * 40];   // [n][k]  stride 40

    const int tid  = threadIdx.x;
    const int wave = tid >> 5, lane = tid & 31;
    const int wm = wave >> 2, wn = wave & 3;          // 2 x 4 wave grid
    const int lr = lane & 15, lh = lane >> 4;
    const int m0 = blockIdx.x * 64;

    const int arow = tid >> 2, aq = tid & 3;          // A staging assignment
    const int gr  = m0 + arow;
    const int grc = (gr < M) ? gr : (M - 1);          // clamped row (always valid)
    const bool rok = (gr < M);
    const float*    aprow = A  + (size_t)grc * lda;
    const _Float16* bprow = Wt + (size_t)tid * K;

    v8f acc[2][4];
    #pragma unroll
    for (int i = 0; i < 2; i++)
        #pragma unroll
        for (int j = 0; j < 4; j++)
            #pragma unroll
            for (int r = 0; r < 8; r++) acc[i][j][r] = 0.f;

    float ar[8];           // double-buffered A chunk (masked, f32)

    auto loadA = [&](int k0) {
        const int kk = k0 + aq * 8;
        if (vec_ok) {
            // lda % 4 == 0 and K == lda: kk+8 <= lda always; only row needs masking
            const float4* ap = (const float4*)(aprow + kk);
            float4 f0 = ap[0];
            float4 f1 = ap[1];
            ar[0] = rok ? f0.x : 0.f; ar[1] = rok ? f0.y : 0.f;
            ar[2] = rok ? f0.z : 0.f; ar[3] = rok ? f0.w : 0.f;
            ar[4] = rok ? f1.x : 0.f; ar[5] = rok ? f1.y : 0.f;
            ar[6] = rok ? f1.z : 0.f; ar[7] = rok ? f1.w : 0.f;
        } else {
            #pragma unroll
            for (int j = 0; j < 8; j++) {
                int kj  = kk + j;
                int kjc = (kj < lda) ? kj : (lda - 1);   // always-valid address
                float f = aprow[kjc];
                ar[j] = (rok && kj < lda) ? f : 0.f;
            }
        }
    };

    loadA(0);

    for (int k0 = 0; k0 < K; k0 += 32) {
        // --- stage B (short register liveness: load then store immediately) ---
        {
            const float4* s = (const float4*)(bprow + k0);
            float4 b0 = s[0], b1 = s[1], b2 = s[2], b3 = s[3];
            float4* dp = (float4*)&Bs[tid * 40];
            dp[0] = b0; dp[1] = b1; dp[2] = b2; dp[3] = b3;
        }
        // --- stage A from double-buffer registers ---
        {
            _Float16* dstp = &As[arow * 40 + aq * 8];
            #pragma unroll
            for (int j = 0; j < 8; j++) dstp[j] = (_Float16)ar[j];
        }
        __syncthreads();

        // issue next A chunk's global loads before compute (latency hidden behind WMMAs)
        if (k0 + 32 < K) loadA(k0 + 32);

        // --- A fragments (kept live; 16 VGPRs) ---
        v16h af[2];
        #pragma unroll
        for (int i = 0; i < 2; i++) {
            int r = wm * 32 + i * 16 + lr;            // A row = lane&15
            v8h a0 = *(const v8h*)&As[r * 40 + lh * 8];        // K = lh*8 .. +7
            v8h a1 = *(const v8h*)&As[r * 40 + 16 + lh * 8];   // K = 16+lh*8 .. +7
            af[i] = __builtin_shufflevector(a0, a1, 0,1,2,3,4,5,6,7,8,9,10,11,12,13,14,15);
        }
        // --- B fragments one N-tile at a time (8 VGPRs transient) ---
        #pragma unroll
        for (int j = 0; j < 4; j++) {
            int n = wn * 64 + j * 16 + lr;            // B col = lane&15
            v8h b0 = *(const v8h*)&Bs[n * 40 + lh * 8];        // K = lh*8 .. +7
            v8h b1 = *(const v8h*)&Bs[n * 40 + 16 + lh * 8];   // K = 16+lh*8 .. +7
            v16h bf = __builtin_shufflevector(b0, b1, 0,1,2,3,4,5,6,7,8,9,10,11,12,13,14,15);
            acc[0][j] = __builtin_amdgcn_wmma_f32_16x16x32_f16(
                false, af[0], false, bf, (short)0, acc[0][j], false, false);
            acc[1][j] = __builtin_amdgcn_wmma_f32_16x16x32_f16(
                false, af[1], false, bf, (short)0, acc[1][j], false, false);
        }
        __syncthreads();
    }

    // --- epilogue: D layout row = base + lh*8 + r, col = base + (lane&15) ---
    if (m0 + 64 <= M) {
        // full tile: no row guards
        #pragma unroll
        for (int i = 0; i < 2; i++) {
            #pragma unroll
            for (int j = 0; j < 4; j++) {
                int col = wn * 64 + j * 16 + lr;
                float badd = bias ? bias[col] : 0.f;
                #pragma unroll
                for (int r = 0; r < 8; r++) {
                    int row = m0 + wm * 32 + i * 16 + lh * 8 + r;
                    float v = acc[i][j][r] + badd;
                    if (do_relu) v = fmaxf(v, 0.f);
                    C[(size_t)row * 256 + col] = v;
                }
            }
        }
    } else {
        #pragma unroll
        for (int i = 0; i < 2; i++) {
            #pragma unroll
            for (int j = 0; j < 4; j++) {
                int col = wn * 64 + j * 16 + lr;
                float badd = bias ? bias[col] : 0.f;
                #pragma unroll
                for (int r = 0; r < 8; r++) {
                    int row = m0 + wm * 32 + i * 16 + lh * 8 + r;
                    if (row < M) {
                        float v = acc[i][j][r] + badd;
                        if (do_relu) v = fmaxf(v, 0.f);
                        C[(size_t)row * 256 + col] = v;
                    }
                }
            }
        }
    }
}

// ---------------- degree + loop-attr accumulation ----------------
__global__ void deg_kernel(const int* __restrict__ dst, const float* __restrict__ eattr,
                           float* __restrict__ deg, float* __restrict__ lattr) {
    int e = blockIdx.x * blockDim.x + threadIdx.x;
    if (e >= EE) return;
    int d = dst[e];
    atomicAdd(&deg[d], 1.f);
    #pragma unroll
    for (int j = 0; j < DEA; j++)
        atomicAdd(&lattr[(size_t)d * DEA + j], eattr[(size_t)e * DEA + j]);
}
__global__ void loopdiv_kernel(const float* __restrict__ deg, float* __restrict__ lattr) {
    int n = blockIdx.x * blockDim.x + threadIdx.x;
    if (n >= NN) return;
    float inv = 1.f / fmaxf(deg[n], 1.f);
    #pragma unroll
    for (int j = 0; j < DEA; j++) lattr[(size_t)n * DEA + j] *= inv;
}

// ---------------- W_eff[d][h] = sum_c edge_W[d, h*64+c] * att_edge[h, c] ----------------
__global__ void weff_kernel(const float* __restrict__ eW, const float* __restrict__ aE,
                            float* __restrict__ weff) {
    int t = threadIdx.x;
    if (t >= DEA * HEADS) return;
    int d = t >> 2, h = t & 3;
    float s = 0.f;
    for (int c = 0; c < CH; c++) s += eW[d * 256 + h * 64 + c] * aE[h * 64 + c];
    weff[d * HEADS + h] = s;
}

// ---------------- a_s / a_d: one wave per node, 8 channels per lane ----------------
__global__ void asd_kernel(const float* __restrict__ xh,
                           const float* __restrict__ att_s, const float* __restrict__ att_d,
                           float* __restrict__ a_s, float* __restrict__ a_d) {
    int gtid = blockIdx.x * blockDim.x + threadIdx.x;
    int n = gtid >> 5;
    if (n >= NN) return;
    int lane = gtid & 31;
    int c0 = lane * 8, head = lane >> 3;
    const float* xp = xh + (size_t)n * 256 + c0;
    const float* sp = att_s + head * 64 + (lane & 7) * 8;
    const float* dp = att_d + head * 64 + (lane & 7) * 8;
    float s = 0.f, d = 0.f;
    #pragma unroll
    for (int j = 0; j < 8; j++) { float v = xp[j]; s += v * sp[j]; d += v * dp[j]; }
    #pragma unroll
    for (int m = 1; m < 8; m <<= 1) { s += __shfl_xor(s, m, 32); d += __shfl_xor(d, m, 32); }
    if ((lane & 7) == 0) { a_s[n * 4 + head] = s; a_d[n * 4 + head] = d; }
}

// ---------------- logits + segment max (ordered-uint atomicMax) ----------------
__global__ void edge_logits_kernel(const int* __restrict__ src, const int* __restrict__ dst,
                                   const float* __restrict__ eattr, const float* __restrict__ lattr,
                                   const float* __restrict__ a_s, const float* __restrict__ a_d,
                                   const float* __restrict__ weff,
                                   float* __restrict__ logits, unsigned* __restrict__ m_u) {
    int e = blockIdx.x * blockDim.x + threadIdx.x;
    if (e >= ENL) return;
    int s, d; const float* attr;
    if (e < EE) { s = src[e]; d = dst[e]; attr = eattr + (size_t)e * DEA; }
    else        { s = d = e - EE;          attr = lattr + (size_t)(e - EE) * DEA; }
    float ae[HEADS] = {0.f, 0.f, 0.f, 0.f};
    #pragma unroll
    for (int j = 0; j < DEA; j++) {
        float av = attr[j];
        #pragma unroll
        for (int h = 0; h < HEADS; h++) ae[h] += av * weff[j * HEADS + h];
    }
    #pragma unroll
    for (int h = 0; h < HEADS; h++) {
        float x = a_s[s * 4 + h] + a_d[d * 4 + h] + ae[h];
        x = (x > 0.f) ? x : NEG_SLOPE * x;          // leaky relu
        logits[(size_t)e * 4 + h] = x;
        atomicMax(&m_u[d * 4 + h], f2ord(x));
    }
}

// ---------------- exp + denom ----------------
__global__ void edge_exp_kernel(const int* __restrict__ dst, float* __restrict__ logits,
                                const unsigned* __restrict__ m_u, float* __restrict__ denom) {
    int idx = blockIdx.x * blockDim.x + threadIdx.x;
    if (idx >= ENL * 4) return;
    int e = idx >> 2, h = idx & 3;
    int d = (e < EE) ? dst[e] : (e - EE);
    float ex = __expf(logits[idx] - ord2f(m_u[d * 4 + h]));
    logits[idx] = ex;                                // overwrite logits with ex
    atomicAdd(&denom[d * 4 + h], ex);
}

// ---------------- weighted scatter aggregation ----------------
__global__ void aggregate_kernel(const int* __restrict__ src, const int* __restrict__ dst,
                                 const float* __restrict__ ex, const float* __restrict__ denom,
                                 const float* __restrict__ xh, float* __restrict__ out) {
    int idx = blockIdx.x * blockDim.x + threadIdx.x;   // ENL * 64 (float4 per thread)
    if (idx >= ENL * 64) return;
    int e = idx >> 6, q = idx & 63, h = q >> 4;
    int s, d;
    if (e < EE) { s = src[e]; d = dst[e]; } else { s = d = e - EE; }
    float alpha = ex[(size_t)e * 4 + h] / denom[d * 4 + h];
    const float4 v = *(const float4*)(xh + (size_t)s * 256 + q * 4);
    float* o = out + (size_t)d * 256 + q * 4;
    atomicAdd(o + 0, v.x * alpha);
    atomicAdd(o + 1, v.y * alpha);
    atomicAdd(o + 2, v.z * alpha);
    atomicAdd(o + 3, v.w * alpha);
}

// ---------------- batchnorm stats / finalize / apply ----------------
__global__ void bn_stats_kernel(const float* __restrict__ h, float* __restrict__ sum,
                                float* __restrict__ sumsq) {
    int c = threadIdx.x;                  // 256 threads: one channel each
    float s = 0.f, s2 = 0.f;
    for (int r = blockIdx.x; r < NN; r += gridDim.x) {
        float v = h[(size_t)r * 256 + c];
        s += v; s2 += v * v;
    }
    atomicAdd(&sum[c], s);
    atomicAdd(&sumsq[c], s2);
}
__global__ void bn_final_kernel(const float* __restrict__ sum, const float* __restrict__ sumsq,
                                const float* __restrict__ gamma, const float* __restrict__ beta,
                                float* __restrict__ scale, float* __restrict__ shift) {
    int c = threadIdx.x;
    float mu = sum[c] / (float)NN;
    float var = fmaxf(sumsq[c] / (float)NN - mu * mu, 0.f);
    float sc = gamma[c] * rsqrtf(var + BN_EPS);
    scale[c] = sc;
    shift[c] = beta[c] - mu * sc;
}
__global__ void bn_apply_kernel(const float* __restrict__ agg, const float* __restrict__ scale,
                                const float* __restrict__ shift, const float* __restrict__ res,
                                float* __restrict__ out) {
    int idx = blockIdx.x * blockDim.x + threadIdx.x;   // NN * 64
    if (idx >= NN * 64) return;
    int n = idx >> 6, q = idx & 63;
    const float4 o  = *(const float4*)(agg + (size_t)n * 256 + q * 4);
    const float4 sc = *(const float4*)(scale + q * 4);
    const float4 sh = *(const float4*)(shift + q * 4);
    float4 r;
    r.x = fmaxf(o.x * sc.x + sh.x, 0.f);
    r.y = fmaxf(o.y * sc.y + sh.y, 0.f);
    r.z = fmaxf(o.z * sc.z + sh.z, 0.f);
    r.w = fmaxf(o.w * sc.w + sh.w, 0.f);
    if (res) {
        const float4 h0 = *(const float4*)(res + (size_t)n * 256 + q * 4);
        r.x += h0.x; r.y += h0.y; r.z += h0.z; r.w += h0.w;
    }
    *(float4*)(out + (size_t)n * 256 + q * 4) = r;
}

// ---------------- pooling ----------------
__global__ void pool_cnt_kernel(const int* __restrict__ bidx, float* __restrict__ cnt) {
    int n = blockIdx.x * blockDim.x + threadIdx.x;
    if (n < NN) atomicAdd(&cnt[bidx[n]], 1.f);
}
__global__ void pool_sum_kernel(const int* __restrict__ bidx, const float* __restrict__ h,
                                float* __restrict__ g) {
    int idx = blockIdx.x * blockDim.x + threadIdx.x;   // NN * 64
    if (idx >= NN * 64) return;
    int n = idx >> 6, q = idx & 63;
    int b = bidx[n];
    const float4 v = *(const float4*)(h + (size_t)n * 256 + q * 4);
    float* o = g + (size_t)b * 256 + q * 4;
    atomicAdd(o + 0, v.x);
    atomicAdd(o + 1, v.y);
    atomicAdd(o + 2, v.z);
    atomicAdd(o + 3, v.w);
}
__global__ void pool_div_kernel(const float* __restrict__ cnt, float* __restrict__ g) {
    int idx = blockIdx.x * blockDim.x + threadIdx.x;   // BB * 64
    if (idx >= BB * 64) return;
    int b = idx >> 6, q = idx & 63;
    float inv = 1.f / fmaxf(cnt[b], 1.f);
    float4* p = (float4*)(g + (size_t)b * 256 + q * 4);
    float4 v = *p;
    v.x *= inv; v.y *= inv; v.z *= inv; v.w *= inv;
    *p = v;
}

// ---------------- small FC: out[M,On] = act(in[M,K] @ W[K,On] + b) ----------------
__global__ void fc_kernel(const float* __restrict__ in, const float* __restrict__ W,
                          const float* __restrict__ b, float* __restrict__ out,
                          int M, int K, int On, int do_relu) {
    int idx = blockIdx.x * blockDim.x + threadIdx.x;
    if (idx >= M * On) return;
    int row = idx / On, col = idx % On;
    float s = b[col];
    for (int k = 0; k < K; k++) s += in[(size_t)row * K + k] * W[(size_t)k * On + col];
    if (do_relu) s = fmaxf(s, 0.f);
    out[(size_t)row * On + col] = s;
}

// =======================================================================
extern "C" void kernel_launch(void* const* d_in, const int* in_sizes, int n_in,
                              void* d_out, int out_size, void* d_ws, size_t ws_size,
                              hipStream_t stream) {
    (void)in_sizes; (void)n_in; (void)out_size; (void)ws_size;

    const float* x         = (const float*)d_in[0];
    const int*   eidx      = (const int*)d_in[1];
    const int*   src       = eidx;
    const int*   dst       = eidx + EE;
    const float* edge_attr = (const float*)d_in[2];
    const int*   batch_idx = (const int*)d_in[3];
    // d_in[4] = batch_size (constant 4096)
    const float* in_W      = (const float*)d_in[5];
    const float* in_b      = (const float*)d_in[6];
    const float* gat_W     = (const float*)d_in[7];
    const float* att_src   = (const float*)d_in[8];
    const float* att_dst   = (const float*)d_in[9];
    const float* edge_W    = (const float*)d_in[10];
    const float* att_edge  = (const float*)d_in[11];
    // d_in[12] = gat_b : cancels exactly through BatchNorm -> skipped
    const float* bn_gamma  = (const float*)d_in[13];
    const float* bn_beta   = (const float*)d_in[14];
    const float* p_W1      = (const float*)d_in[15];
    const float* p_b1      = (const float*)d_in[16];
    const float* p_W2      = (const float*)d_in[17];
    const float* p_b2      = (const float*)d_in[18];
    const float* p_W3      = (const float*)d_in[19];
    const float* p_b3      = (const float*)d_in[20];
    float* outp            = (float*)d_out;

    // -------- workspace carve-up (256B aligned) --------
    char* base = (char*)d_ws;
    size_t off = 0;
    auto alloc = [&](size_t bytes) -> char* {
        char* p = base + off;
        off += (bytes + 255) & ~(size_t)255;
        return p;
    };
    float*    buf0   = (float*)alloc((size_t)NN * HID * 4);
    float*    buf1   = (float*)alloc((size_t)NN * HID * 4);
    float*    buf2   = (float*)alloc((size_t)NN * HID * 4);
    float*    a_s    = (float*)alloc((size_t)NN * 4 * 4);
    float*    a_d    = (float*)alloc((size_t)NN * 4 * 4);
    float*    exb    = (float*)alloc((size_t)ENL * 4 * 4);
    float*    denom  = (float*)alloc((size_t)NN * 4 * 4);
    unsigned* m_u    = (unsigned*)alloc((size_t)NN * 4 * 4);
    float*    deg    = (float*)alloc((size_t)NN * 4);
    float*    lattr  = (float*)alloc((size_t)NN * DEA * 4);
    float*    bnsum  = (float*)alloc(256 * 4);
    float*    bnss   = (float*)alloc(256 * 4);
    float*    bnsc   = (float*)alloc(256 * 4);
    float*    bnsh   = (float*)alloc(256 * 4);
    float*    weff   = (float*)alloc(DEA * HEADS * 4);
    float*    g      = (float*)alloc((size_t)BB * 256 * 4);
    float*    cnt    = (float*)alloc((size_t)BB * 4);
    float*    z1     = (float*)alloc((size_t)BB * 128 * 4);
    float*    z2     = (float*)alloc((size_t)BB * 64 * 4);
    _Float16* wtproj = (_Float16*)alloc((size_t)256 * KPROJ * 2);
    _Float16* wtgat  = (_Float16*)alloc((size_t)LAY * 256 * 256 * 2);

    auto zero = [&](float* p, size_t nf) {
        int n4 = (int)(nf / 4);
        zero_kernel<<<CDIV(n4, 256), 256, 0, stream>>>(p, n4);
    };

    // -------- one-time per call: weight conversion (f32 -> transposed f16) --------
    cvt_wt_kernel<<<CDIV(256 * KPROJ, 256), 256, 0, stream>>>(in_W, wtproj, DIN, KPROJ);
    for (int l = 0; l < LAY; l++)
        cvt_wt_kernel<<<CDIV(256 * 256, 256), 256, 0, stream>>>(
            gat_W + (size_t)l * 256 * 256, wtgat + (size_t)l * 256 * 256, 256, 256);

    // -------- self-loop edge attrs (mean of incoming) --------
    zero(deg, NN);
    zero(lattr, (size_t)NN * DEA);
    deg_kernel<<<CDIV(EE, 256), 256, 0, stream>>>(dst, edge_attr, deg, lattr);
    loopdiv_kernel<<<CDIV(NN, 256), 256, 0, stream>>>(deg, lattr);

    // -------- input projection: h = relu(x @ in_W + in_b)  (lda=155 -> scalar path) --------
    gemm_f16_wmma<<<CDIV(NN, 64), 256, 0, stream>>>(x, DIN, NN, wtproj, KPROJ, buf0, in_b, 1, 0);

    float* hcur = buf0;   // current node features
    float* hx   = buf1;   // xh / next-h buffer
    float* agg  = buf2;   // aggregation buffer

    for (int l = 0; l < LAY; l++) {
        weff_kernel<<<1, 32, 0, stream>>>(edge_W + (size_t)l * DEA * 256,
                                          att_edge + (size_t)l * HEADS * CH, weff);

        // xh = h @ gat_W[l]  (lda=256 -> vector staging path)
        gemm_f16_wmma<<<CDIV(NN, 64), 256, 0, stream>>>(
            hcur, 256, NN, wtgat + (size_t)l * 256 * 256, 256, hx, nullptr, 0, 1);

        asd_kernel<<<CDIV(NN * 32, 256), 256, 0, stream>>>(
            hx, att_src + (size_t)l * HEADS * CH, att_dst + (size_t)l * HEADS * CH, a_s, a_d);

        zero((float*)m_u, (size_t)NN * 4);      // encoded 0 < any encoded real
        zero(denom, (size_t)NN * 4);
        edge_logits_kernel<<<CDIV(ENL, 256), 256, 0, stream>>>(
            src, dst, edge_attr, lattr, a_s, a_d, weff, exb, m_u);
        edge_exp_kernel<<<CDIV(ENL * 4, 256), 256, 0, stream>>>(dst, exb, m_u, denom);

        zero(agg, (size_t)NN * HID);
        aggregate_kernel<<<CDIV(ENL * 64, 256), 256, 0, stream>>>(src, dst, exb, denom, hx, agg);

        // BatchNorm (batch stats) + relu + residual (gat_b cancels in BN)
        zero(bnsum, 256); zero(bnss, 256);
        bn_stats_kernel<<<1184, 256, 0, stream>>>(agg, bnsum, bnss);
        bn_final_kernel<<<1, 256, 0, stream>>>(bnsum, bnss, bn_gamma + (size_t)l * 256,
                                               bn_beta + (size_t)l * 256, bnsc, bnsh);
        bn_apply_kernel<<<CDIV(NN * 64, 256), 256, 0, stream>>>(
            agg, bnsc, bnsh, (l > 0) ? hcur : nullptr, hx);

        // rotate: new h lives in hx
        float* t = hcur; hcur = hx; hx = t;
    }

    // -------- global mean pool --------
    zero(g, (size_t)BB * 256);
    zero(cnt, BB);
    pool_cnt_kernel<<<CDIV(NN, 256), 256, 0, stream>>>(batch_idx, cnt);
    pool_sum_kernel<<<CDIV(NN * 64, 256), 256, 0, stream>>>(batch_idx, hcur, g);
    pool_div_kernel<<<CDIV(BB * 64, 256), 256, 0, stream>>>(cnt, g);

    // -------- predictor MLP --------
    fc_kernel<<<CDIV(BB * 128, 256), 256, 0, stream>>>(g,  p_W1, p_b1, z1,  BB, 256, 128, 1);
    fc_kernel<<<CDIV(BB * 64, 256), 256, 0, stream>>>(z1, p_W2, p_b2, z2,  BB, 128, 64, 1);
    fc_kernel<<<CDIV(BB, 256), 256, 0, stream>>>(z2, p_W3, p_b3, outp, BB, 64, 1, 0);
}
